// NeuronGptOssExperts_25546465476997
// MI455X (gfx1250) — matrix operand order repaired and checked by reading, not compiled, not run on hardware
//
#include <hip/hip_runtime.h>

#define GELU_ALPHA 1.702f
#define GLU_LIMIT  7.0f

typedef __attribute__((ext_vector_type(16))) __bf16       v16bf;
typedef __attribute__((ext_vector_type(8)))  float        v8f;
typedef __attribute__((ext_vector_type(4)))  unsigned int u32x4;
typedef __attribute__((ext_vector_type(4)))  float        f32x4;

struct FragP { u32x4 lo; u32x4 hi; };
union  FragU { FragP p; v16bf v; };

static constexpr int PA = 72;   // LDS pitch (bf16 elems) for A tiles (row-major [m][k])
static constexpr int PB = 72;   // LDS pitch for B tiles (transposed [n][k])

// A fragment: 16x32 bf16. ISA 7.12.2: lanes 0-15 -> M=lane, K chunks [0..8)+[16..24);
// lanes 16-31 -> M=lane-16, K chunks [8..16)+[24..32).
__device__ __forceinline__ v16bf load_a_frag(const __bf16* rowbase, int lane, int ks) {
    int m  = lane & 15;
    int k0 = ks + ((lane & 16) ? 8 : 0);
    const __bf16* p = rowbase + m * PA + k0;
    FragU f;
    f.p.lo = *(const u32x4*)(p);
    f.p.hi = *(const u32x4*)(p + 16);
    return f.v;
}

// B fragment: 32x16 bf16 from transposed LDS panel [n][k]. Lanes 0-15 -> N=lane, K chunks [0..8)+[16..24);
// lanes 16-31 -> N=lane-16, K chunks [8..16)+[24..32).
__device__ __forceinline__ v16bf load_b_frag(const __bf16* nbase, int lane, int ks) {
    int n  = lane & 15;
    int k0 = ks + ((lane & 16) ? 8 : 0);
    const __bf16* p = nbase + n * PB + k0;
    FragU f;
    f.p.lo = *(const u32x4*)(p);
    f.p.hi = *(const u32x4*)(p + 16);
    return f.v;
}

#define WMMA_BF16(A, B, C) \
    __builtin_amdgcn_wmma_f32_16x16x32_bf16(false, (A), false, (B), (short)0, (C), false, false)

// Pack two floats into one dword of bf16 pairs (low addr -> low 16 bits).
__device__ __forceinline__ unsigned pack_bf16(float a, float b) {
    unsigned short ua = __builtin_bit_cast(unsigned short, (__bf16)a);
    unsigned short ub = __builtin_bit_cast(unsigned short, (__bf16)b);
    return ((unsigned)ub << 16) | ua;
}

// gfx1250 async copy: global -> LDS, 16B per lane, tracked on ASYNCcnt.
__device__ __forceinline__ void async_copy16(void* lds_dst, const void* gsrc) {
    unsigned lds = (unsigned)(uintptr_t)lds_dst;   // low 32 bits of generic = LDS offset
    asm volatile("global_load_async_to_lds_b128 %0, %1, off"
                 :: "v"(lds), "v"((unsigned long long)(uintptr_t)gsrc)
                 : "memory");
}
__device__ __forceinline__ void async_wait0() {
    asm volatile("s_wait_asynccnt 0x0" ::: "memory");
}

// ---------------- x f32 -> bf16 ----------------
__global__ __launch_bounds__(256) void cvt_bf16(const float* __restrict__ in,
                                                __bf16* __restrict__ out, int n) {
    int i = (blockIdx.x * 256 + threadIdx.x) * 4;
    if (i + 3 < n) {
        f32x4 v = *(const f32x4*)(in + i);
        out[i]     = (__bf16)v.x;
        out[i + 1] = (__bf16)v.y;
        out[i + 2] = (__bf16)v.z;
        out[i + 3] = (__bf16)v.w;
    }
}

// ---------------- GEMM1 + clamped-SiLU-GLU -> act[e][t][f] (bf16) ----------------
// Block: 128 thr (4 waves). Tile: 128 rows x 64 f (=>128 gate/up cols), K staged 64.
// Each wave owns 32 rows (2 A-frags) so every B fragment feeds 2 WMMAs.
__global__ __launch_bounds__(128) void gemm1_act(const __bf16* __restrict__ xbf,
                                                 const float* __restrict__ w1,
                                                 const float* __restrict__ b1,
                                                 __bf16* __restrict__ act) {
    __shared__ __bf16 As[128 * PA];
    __shared__ __bf16 Bg[64 * PB];   // gate columns (even c), transposed [f][k]
    __shared__ __bf16 Bu[64 * PB];   // up   columns (odd  c), transposed [f][k]

    const int e    = blockIdx.z;
    const int f0   = blockIdx.x * 64;
    const int t0   = blockIdx.y * 128;
    const int tid  = threadIdx.x;
    const int wave = tid >> 5;
    const int lane = tid & 31;

    v8f zero = {0.f, 0.f, 0.f, 0.f, 0.f, 0.f, 0.f, 0.f};
    v8f accg[2][4], accu[2][4];
#pragma unroll
    for (int mi = 0; mi < 2; ++mi)
#pragma unroll
        for (int j = 0; j < 4; ++j) { accg[mi][j] = zero; accu[mi][j] = zero; }

    for (int kb = 0; kb < 1024; kb += 64) {
        __syncthreads();
        {   // stage A (async DMA to LDS): 128 rows x 64 k bf16, one row per thread
            const __bf16* src = xbf + (size_t)(t0 + tid) * 1024 + kb;
            __bf16* dst = As + tid * PA;
#pragma unroll
            for (int i = 0; i < 8; ++i)
                async_copy16(dst + i * 8, src + i * 8);
        }
        {   // stage B: 64k x 128c f32 -> de-interleave gate/up, transpose, pack k-pairs
            int kp = tid & 31;        // k-pair 0..31 -> rows k, k+1
            int cg = tid >> 2 >> 3;   // 0..3 (tid>>5), 32 c-columns (16 f) each
            int k  = 2 * kp;
            const float* s0 = w1 + ((size_t)e * 1024 + kb + k) * 4096 + 2 * f0 + cg * 32;
            const float* s1 = s0 + 4096;
#pragma unroll
            for (int i = 0; i < 8; ++i) {
                f32x4 w0 = *(const f32x4*)(s0 + i * 4);
                f32x4 w1v = *(const f32x4*)(s1 + i * 4);
                int fl = cg * 16 + i * 2;
                *(unsigned*)(Bg + fl * PB + k)       = pack_bf16(w0.x, w1v.x);
                *(unsigned*)(Bu + fl * PB + k)       = pack_bf16(w0.y, w1v.y);
                *(unsigned*)(Bg + (fl + 1) * PB + k) = pack_bf16(w0.z, w1v.z);
                *(unsigned*)(Bu + (fl + 1) * PB + k) = pack_bf16(w0.w, w1v.w);
            }
        }
        async_wait0();
        __syncthreads();
#pragma unroll
        for (int ks = 0; ks < 64; ks += 32) {
            v16bf a0 = load_a_frag(As + (wave * 32) * PA, lane, ks);
            v16bf a1 = load_a_frag(As + (wave * 32 + 16) * PA, lane, ks);
#pragma unroll
            for (int j = 0; j < 4; ++j) {
                v16bf bg = load_b_frag(Bg + j * 16 * PB, lane, ks);
                accg[0][j] = WMMA_BF16(a0, bg, accg[0][j]);
                accg[1][j] = WMMA_BF16(a1, bg, accg[1][j]);
                v16bf bu = load_b_frag(Bu + j * 16 * PB, lane, ks);
                accu[0][j] = WMMA_BF16(a0, bu, accu[0][j]);
                accu[1][j] = WMMA_BF16(a1, bu, accu[1][j]);
            }
        }
    }

    // Epilogue: bias + clamp + SiLU-GLU, store bf16 act.
    const int col   = lane & 15;
    const int rbase = (lane & 16) ? 8 : 0;
#pragma unroll
    for (int j = 0; j < 4; ++j) {
        int f = f0 + j * 16 + col;
        float bg = b1[(size_t)e * 4096 + 2 * f];
        float bu = b1[(size_t)e * 4096 + 2 * f + 1];
#pragma unroll
        for (int mi = 0; mi < 2; ++mi) {
#pragma unroll
            for (int r = 0; r < 8; ++r) {
                float g = accg[mi][j][r] + bg;
                float u = accu[mi][j][r] + bu;
                g = fminf(g, GLU_LIMIT);
                u = fminf(fmaxf(u, -GLU_LIMIT), GLU_LIMIT);
                float glu = g / (1.0f + __expf(-GELU_ALPHA * g));
                float a   = (u + 1.0f) * glu;
                int t = t0 + wave * 32 + mi * 16 + r + rbase;
                act[((size_t)e * 2048 + t) * 2048 + f] = (__bf16)a;
            }
        }
    }
}

// ---------------- GEMM2: out[t][h] = sum_e rw[t,e]*(act_e . W2_e + b2_e) ----------------
// Block: 128 thr (4 waves). Tile: 128 rows x 64 h. Experts looped inside => no atomics.
__global__ __launch_bounds__(128) void gemm2_moe(const __bf16* __restrict__ act,
                                                 const float* __restrict__ w2,
                                                 const float* __restrict__ b2,
                                                 const float* __restrict__ routing,
                                                 float* __restrict__ out) {
    __shared__ __bf16 As[128 * PA];
    __shared__ __bf16 Bh[64 * PB];
    __shared__ float  rw[128 * 16];

    const int h0   = blockIdx.x * 64;
    const int t0   = blockIdx.y * 128;
    const int tid  = threadIdx.x;
    const int wave = tid >> 5;
    const int lane = tid & 31;

    for (int i = tid; i < 128 * 16; i += 128)
        rw[i] = routing[(size_t)t0 * 16 + i];

    v8f zero = {0.f, 0.f, 0.f, 0.f, 0.f, 0.f, 0.f, 0.f};
    v8f facc[2][4];
#pragma unroll
    for (int mi = 0; mi < 2; ++mi)
#pragma unroll
        for (int j = 0; j < 4; ++j) facc[mi][j] = zero;

    const int rbase = (lane & 16) ? 8 : 0;

    for (int e = 0; e < 16; ++e) {
        v8f tmp[2][4];
#pragma unroll
        for (int mi = 0; mi < 2; ++mi)
#pragma unroll
            for (int j = 0; j < 4; ++j) tmp[mi][j] = zero;

        for (int kb = 0; kb < 2048; kb += 64) {
            __syncthreads();
            {   // stage A (async DMA to LDS): act rows, one row per thread
                const __bf16* src = act + ((size_t)e * 2048 + t0 + tid) * 2048 + kb;
                __bf16* dst = As + tid * PA;
#pragma unroll
                for (int i = 0; i < 8; ++i)
                    async_copy16(dst + i * 8, src + i * 8);
            }
            {   // stage B: W2 64k x 64h f32 -> transpose + pack k-pairs
                int kp = tid & 31;
                int hg = tid >> 5;    // 0..3, 16 h each
                int k  = 2 * kp;
                const float* s0 = w2 + ((size_t)e * 2048 + kb + k) * 1024 + h0 + hg * 16;
                const float* s1 = s0 + 1024;
#pragma unroll
                for (int i = 0; i < 4; ++i) {
                    f32x4 w0 = *(const f32x4*)(s0 + i * 4);
                    f32x4 w1v = *(const f32x4*)(s1 + i * 4);
#pragma unroll
                    for (int c = 0; c < 4; ++c)
                        *(unsigned*)(Bh + (hg * 16 + i * 4 + c) * PB + k) = pack_bf16(w0[c], w1v[c]);
                }
            }
            async_wait0();
            __syncthreads();
#pragma unroll
            for (int ks = 0; ks < 64; ks += 32) {
                v16bf a0 = load_a_frag(As + (wave * 32) * PA, lane, ks);
                v16bf a1 = load_a_frag(As + (wave * 32 + 16) * PA, lane, ks);
#pragma unroll
                for (int j = 0; j < 4; ++j) {
                    v16bf bh = load_b_frag(Bh + j * 16 * PB, lane, ks);
                    tmp[0][j] = WMMA_BF16(a0, bh, tmp[0][j]);
                    tmp[1][j] = WMMA_BF16(a1, bh, tmp[1][j]);
                }
            }
        }
        // per-row routing scale into final accumulators (C layout: row = r + (lane>=16?8:0))
#pragma unroll
        for (int mi = 0; mi < 2; ++mi)
#pragma unroll
            for (int r = 0; r < 8; ++r) {
                float rv = rw[(wave * 32 + mi * 16 + r + rbase) * 16 + e];
#pragma unroll
                for (int j = 0; j < 4; ++j) facc[mi][j][r] += rv * tmp[mi][j][r];
            }
    }

    // Epilogue: add routed down-bias, store f32.
    const int col = lane & 15;
#pragma unroll
    for (int mi = 0; mi < 2; ++mi) {
#pragma unroll
        for (int j = 0; j < 4; ++j) {
            int h = h0 + j * 16 + col;
#pragma unroll
            for (int r = 0; r < 8; ++r) {
                int rl = wave * 32 + mi * 16 + r + rbase;
                float bias = 0.0f;
#pragma unroll
                for (int e = 0; e < 16; ++e)
                    bias += rw[rl * 16 + e] * b2[(size_t)e * 1024 + h];
                out[(size_t)(t0 + rl) * 1024 + h] = facc[mi][j][r] + bias;
            }
        }
    }
}

extern "C" void kernel_launch(void* const* d_in, const int* in_sizes, int n_in,
                              void* d_out, int out_size, void* d_ws, size_t ws_size,
                              hipStream_t stream) {
    const float* hs = (const float*)d_in[0];   // (4,512,1024)
    const float* rt = (const float*)d_in[1];   // (2048,16)
    const float* w1 = (const float*)d_in[2];   // (16,1024,4096)
    const float* b1 = (const float*)d_in[3];   // (16,4096)
    const float* w2 = (const float*)d_in[4];   // (16,2048,1024)
    const float* b2 = (const float*)d_in[5];   // (16,1024)
    float* out = (float*)d_out;                // (4,512,1024)

    __bf16* xbf  = (__bf16*)d_ws;                                    // 4 MB
    __bf16* actb = (__bf16*)((char*)d_ws + (size_t)2048 * 1024 * 2); // 128 MB

    const int nx = 2048 * 1024;
    cvt_bf16<<<nx / (256 * 4), 256, 0, stream>>>(hs, xbf, nx);
    gemm1_act<<<dim3(32, 16, 16), 128, 0, stream>>>(xbf, w1, b1, actb);
    gemm2_moe<<<dim3(16, 16), 128, 0, stream>>>(actb, w2, b2, rt, out);
}
